// GraphRegressionModel_40157944217914
// MI455X (gfx1250) — matrix operand order repaired and checked by reading, not compiled, not run on hardware
//
#include <hip/hip_runtime.h>

#define N_NODES  100000
#define N_EDGES  800000
#define N_GRAPHS 128
#define HID      64

typedef __attribute__((ext_vector_type(16))) _Float16 v16h;
typedef __attribute__((ext_vector_type(8)))  _Float16 v8h;
typedef __attribute__((ext_vector_type(8)))  float    v8f;
typedef __attribute__((ext_vector_type(4)))  float    v4f;

union HFrag { v16h v; v8h h[2]; };

__device__ __forceinline__ v8f wmma16(v16h a, v16h b, v8f c) {
  // D = A(16x32 f16) x B(32x16 f16) + C(16x16 f32)
  return __builtin_amdgcn_wmma_f32_16x16x32_f16(false, a, false, b, (short)0, c, false, false);
}

__device__ __forceinline__ v8f splat8(float x) { v8f r = {x,x,x,x,x,x,x,x}; return r; }

__device__ __forceinline__ v8h cvt8(const float* __restrict__ p) {
  v4f a = *(const v4f*)p;
  v4f b = *(const v4f*)(p + 4);
  v8h r;
  r[0]=(_Float16)a[0]; r[1]=(_Float16)a[1]; r[2]=(_Float16)a[2]; r[3]=(_Float16)a[3];
  r[4]=(_Float16)b[0]; r[5]=(_Float16)b[1]; r[6]=(_Float16)b[2]; r[7]=(_Float16)b[3];
  return r;
}

__device__ __forceinline__ v8h cvt8s(const float* __restrict__ p, float s) {
  v4f a = *(const v4f*)p;
  v4f b = *(const v4f*)(p + 4);
  v8h r;
  r[0]=(_Float16)(a[0]*s); r[1]=(_Float16)(a[1]*s); r[2]=(_Float16)(a[2]*s); r[3]=(_Float16)(a[3]*s);
  r[4]=(_Float16)(b[0]*s); r[5]=(_Float16)(b[1]*s); r[6]=(_Float16)(b[2]*s); r[7]=(_Float16)(b[3]*s);
  return r;
}

// ---------------- K0: zero accumulator workspace ----------------
__global__ void k_zero(float* __restrict__ p, int n) {
  int i = blockIdx.x * blockDim.x + threadIdx.x;
  int stride = gridDim.x * blockDim.x;
  for (; i < n; i += stride) p[i] = 0.0f;
}

// ---------------- K1: h = relu(x @ Wn + bn), store f16 ----------------
__global__ void __launch_bounds__(128)
k_node_embed(const float* __restrict__ x, const float* __restrict__ Wn,
             const float* __restrict__ bn, _Float16* __restrict__ h) {
  int w = threadIdx.x >> 5;
  int lane = threadIdx.x & 31;
  int m = lane & 15, half = lane >> 4;

  // B fragments (Wn: [32][64] f32 row-major) loaded ONCE, reused across tiles
  HFrag b[4];
  float bias[4];
#pragma unroll
  for (int t = 0; t < 4; ++t) {
    int n = t * 16 + m;
    bias[t] = bn[n];
#pragma unroll
    for (int j = 0; j < 16; ++j) {
      int k = (j & 8) * 2 + half * 8 + (j & 7);
      b[t].v[j] = (_Float16)Wn[k * HID + n];
    }
  }

  // persistent waves: grid-stride over 16-row tiles (wave-uniform loop)
  for (int tile = blockIdx.x * 4 + w; tile * 16 < N_NODES; tile += gridDim.x * 4) {
    int row = tile * 16 + m;
    const float* xr = x + (long)row * 32 + half * 8;
    HFrag a;
    a.h[0] = cvt8(xr);
    a.h[1] = cvt8(xr + 16);

#pragma unroll
    for (int t = 0; t < 4; ++t) {
      int n = t * 16 + m;
      v8f c = splat8(bias[t]);          // bias folded into accumulator
      c = wmma16(a.v, b[t].v, c);
#pragma unroll
      for (int r = 0; r < 8; ++r) {
        float v = c[r] > 0.0f ? c[r] : 0.0f;
        int mm = half * 8 + r;
        h[(long)(tile * 16 + mm) * HID + n] = (_Float16)v;
      }
    }
  }
}

// ---------------- K2: edge embed + message MLP + scatter-add ----------------
__global__ void __launch_bounds__(128)
k_edge_msg(const float* __restrict__ edge_attr, const int* __restrict__ eidx,
           const float* __restrict__ We, const float* __restrict__ be,
           const float* __restrict__ WM, const float* __restrict__ bM,
           const _Float16* __restrict__ h, float* __restrict__ aggr,
           float* __restrict__ deg) {
  // Pre-swizzled B fragments: frag line = lane*16 halfs (32B) -> 2x ds_load_b128
  __shared__ __attribute__((aligned(16))) _Float16 wm_lds[24 * 512]; // 6 ksteps x 4 ntiles
  __shared__ __attribute__((aligned(16))) _Float16 we_lds[4 * 512];  // K padded 16->32
  __shared__ __attribute__((aligned(16))) _Float16 estage[4 * 16 * HID]; // per-wave e tile

  int tid = threadIdx.x;
  for (int i = tid; i < 24 * 512; i += blockDim.x) {
    int f = i >> 9, ln = (i >> 4) & 31, j = i & 15;
    int ks = f >> 2, t = f & 3;
    int k = ks * 32 + (j & 8) * 2 + (ln >> 4) * 8 + (j & 7);
    int n = t * 16 + (ln & 15);
    wm_lds[i] = (_Float16)WM[k * HID + n];
  }
  for (int i = tid; i < 4 * 512; i += blockDim.x) {
    int t = i >> 9, ln = (i >> 4) & 31, j = i & 15;
    int k = (j & 8) * 2 + (ln >> 4) * 8 + (j & 7);
    int n = t * 16 + (ln & 15);
    we_lds[i] = (k < 16) ? (_Float16)We[k * HID + n] : (_Float16)0.0f;
  }
  __syncthreads();

  int w = tid >> 5;
  int lane = tid & 31;
  int m = lane & 15, half = lane >> 4;
  _Float16* est = estage + w * (16 * HID);

  float biasE[4], biasM[4];
#pragma unroll
  for (int t = 0; t < 4; ++t) { biasE[t] = be[t * 16 + m]; biasM[t] = bM[t * 16 + m]; }

  // persistent waves: grid-stride over 16-edge tiles (wave-uniform loop)
  for (int tile = blockIdx.x * 4 + w; tile * 16 < N_EDGES; tile += gridDim.x * 4) {
    int e0 = tile * 16;
    int si = eidx[e0 + m];                        // edge_index[0] = src
    int di = eidx[N_EDGES + e0 + m];              // edge_index[1] = dst

    // ---- edge embed: e = relu(edge_attr @ We + be); K padded 16->32 ----
    HFrag a;
    {
      const float* er = edge_attr + (long)(e0 + m) * 16 + half * 8;
      a.h[0] = cvt8(er);
      v8h z = {(_Float16)0.f,(_Float16)0.f,(_Float16)0.f,(_Float16)0.f,
               (_Float16)0.f,(_Float16)0.f,(_Float16)0.f,(_Float16)0.f};
      a.h[1] = z;                                 // padded K=16..31
    }
#pragma unroll
    for (int t = 0; t < 4; ++t) {
      int n = t * 16 + m;
      v8f c = splat8(biasE[t]);
      HFrag b;
      const _Float16* wp = we_lds + t * 512 + lane * 16;
      b.h[0] = *(const v8h*)wp;
      b.h[1] = *(const v8h*)(wp + 8);
      c = wmma16(a.v, b.v, c);
#pragma unroll
      for (int r = 0; r < 8; ++r) {
        float v = c[r] > 0.0f ? c[r] : 0.0f;
        int mm = half * 8 + r;
        est[mm * HID + n] = (_Float16)v;          // row-major f16 stage (same wave reads)
      }
    }

    // ---- message: msg = relu([h[dst] | h[src] | e] @ WM + bM), K=192 ----
    v8f mc[4];
#pragma unroll
    for (int t = 0; t < 4; ++t) mc[t] = splat8(biasM[t]);

    const _Float16* hd = h + (long)di * HID;
    const _Float16* hs = h + (long)si * HID;
#pragma unroll
    for (int ks = 0; ks < 6; ++ks) {
      const _Float16* base =
          (ks < 2) ? (hd + ks * 32)
        : (ks < 4) ? (hs + (ks - 2) * 32)
                   : (est + m * HID + (ks - 4) * 32);
      HFrag af;
      af.h[0] = *(const v8h*)(base + half * 8);
      af.h[1] = *(const v8h*)(base + half * 8 + 16);
#pragma unroll
      for (int t = 0; t < 4; ++t) {
        HFrag bf;
        const _Float16* wp = wm_lds + (ks * 4 + t) * 512 + lane * 16;
        bf.h[0] = *(const v8h*)wp;
        bf.h[1] = *(const v8h*)(wp + 8);
        mc[t] = wmma16(af.v, bf.v, mc[t]);
      }
    }

    // ---- scatter-add msg onto aggr[dst], count degrees ----
    int drow[8];
#pragma unroll
    for (int r = 0; r < 8; ++r) drow[r] = eidx[N_EDGES + e0 + half * 8 + r];
#pragma unroll
    for (int t = 0; t < 4; ++t) {
      int n = t * 16 + m;
#pragma unroll
      for (int r = 0; r < 8; ++r) {
        float v = mc[t][r] > 0.0f ? mc[t][r] : 0.0f;
        atomicAdd(aggr + (long)drow[r] * HID + n, v);
      }
    }
    if (lane < 16) atomicAdd(deg + di, 1.0f);
  }
}

// ---------------- K3: h2 = relu([h | aggr/deg] @ WU + bU) + graph pool ----------------
__global__ void __launch_bounds__(128)
k_update_pool(const _Float16* __restrict__ h, const float* __restrict__ aggr,
              const float* __restrict__ deg, const float* __restrict__ WU,
              const float* __restrict__ bU, const int* __restrict__ batch,
              float* __restrict__ pool, float* __restrict__ gcnt) {
  __shared__ __attribute__((aligned(16))) _Float16 wu_lds[16 * 512]; // 4 ksteps x 4 ntiles
  int tid = threadIdx.x;
  for (int i = tid; i < 16 * 512; i += blockDim.x) {
    int f = i >> 9, ln = (i >> 4) & 31, j = i & 15;
    int ks = f >> 2, t = f & 3;
    int k = ks * 32 + (j & 8) * 2 + (ln >> 4) * 8 + (j & 7);
    int n = t * 16 + (ln & 15);
    wu_lds[i] = (_Float16)WU[k * HID + n];
  }
  __syncthreads();

  int w = tid >> 5;
  int lane = tid & 31;
  int m = lane & 15, half = lane >> 4;

  float biasU[4];
#pragma unroll
  for (int t = 0; t < 4; ++t) biasU[t] = bU[t * 16 + m];

  // persistent waves: grid-stride over 16-node tiles (wave-uniform loop)
  for (int wave = blockIdx.x * 4 + w; wave * 16 < N_NODES; wave += gridDim.x * 4) {
    int node = wave * 16 + m;

    float d = deg[node];
    float invd = 1.0f / (d > 1.0f ? d : 1.0f);    // segment-mean (0-deg rows are all-zero)

    v8f c[4];
#pragma unroll
    for (int t = 0; t < 4; ++t) c[t] = splat8(biasU[t]);

#pragma unroll
    for (int ks = 0; ks < 4; ++ks) {
      HFrag af;
      if (ks < 2) {
        const _Float16* p = h + (long)node * HID + ks * 32 + half * 8;
        af.h[0] = *(const v8h*)p;
        af.h[1] = *(const v8h*)(p + 16);
      } else {
        const float* p = aggr + (long)node * HID + (ks - 2) * 32 + half * 8;
        af.h[0] = cvt8s(p, invd);
        af.h[1] = cvt8s(p + 16, invd);
      }
#pragma unroll
      for (int t = 0; t < 4; ++t) {
        HFrag bf;
        const _Float16* wp = wu_lds + (ks * 4 + t) * 512 + lane * 16;
        bf.h[0] = *(const v8h*)wp;
        bf.h[1] = *(const v8h*)(wp + 8);
        c[t] = wmma16(af.v, bf.v, c[t]);
      }
    }

    int brow[8];
#pragma unroll
    for (int r = 0; r < 8; ++r) brow[r] = batch[wave * 16 + half * 8 + r];
#pragma unroll
    for (int t = 0; t < 4; ++t) {
      int n = t * 16 + m;
#pragma unroll
      for (int r = 0; r < 8; ++r) {
        float v = c[t][r] > 0.0f ? c[t][r] : 0.0f;
        atomicAdd(pool + brow[r] * HID + n, v);
      }
    }
    if (lane < 16) atomicAdd(gcnt + batch[node], 1.0f);
  }
}

// ---------------- K4: out = (pool/cnt) @ Wh + bh ----------------
__global__ void k_final(const float* __restrict__ pool, const float* __restrict__ gcnt,
                        const float* __restrict__ Wh, const float* __restrict__ bh,
                        float* __restrict__ out) {
  int g = threadIdx.x;
  if (g >= N_GRAPHS) return;
  float cnt = gcnt[g];
  float inv = 1.0f / (cnt > 1.0f ? cnt : 1.0f);
  float s = 0.0f;
#pragma unroll
  for (int k = 0; k < HID; ++k) s += pool[g * HID + k] * Wh[k];
  out[g] = s * inv + bh[0];
}

extern "C" void kernel_launch(void* const* d_in, const int* in_sizes, int n_in,
                              void* d_out, int out_size, void* d_ws, size_t ws_size,
                              hipStream_t stream) {
  const float* x         = (const float*)d_in[0];
  const float* edge_attr = (const float*)d_in[1];
  const int*   eidx      = (const int*)d_in[2];   // [2][E]
  const int*   batch     = (const int*)d_in[3];   // [N]
  const float* Wn = (const float*)d_in[4];
  const float* bn = (const float*)d_in[5];
  const float* We = (const float*)d_in[6];
  const float* be = (const float*)d_in[7];
  const float* WM = (const float*)d_in[8];
  const float* bM = (const float*)d_in[9];
  const float* WU = (const float*)d_in[10];
  const float* bU = (const float*)d_in[11];
  const float* Wh = (const float*)d_in[12];
  const float* bh = (const float*)d_in[13];
  float* out = (float*)d_out;

  // Workspace layout (bytes):
  //   [0, 12.8e6)           h  (f16, N x 64)
  //   [12.8e6, 38.4e6)      aggr (f32, N x 64)
  //   [38.4e6, 38.8e6)      deg  (f32, N)
  //   [38.8e6, +32KB)       pool (f32, G x 64)
  //   [.., +512B)           gcnt (f32, G)
  char* ws = (char*)d_ws;
  _Float16* h  = (_Float16*)ws;
  float* aggr  = (float*)(ws + 12800000);
  float* deg   = (float*)(ws + 38400000);
  float* pool  = (float*)(ws + 38800000);
  float* gcnt  = (float*)(ws + 38832768);

  // zero aggr+deg+pool+gcnt (contiguous starting at aggr)
  int zero_n = (25600000 + 400000 + 32768 + 512) / 4;
  k_zero<<<2048, 256, 0, stream>>>(aggr, zero_n);

  // node embed: 6250 tiles of 16 rows; persistent waves (~2.5 tiles/wave)
  k_node_embed<<<625, 128, 0, stream>>>(x, Wn, bn, h);

  // edge embed + message + scatter: 50000 tiles of 16 edges; 10 tiles/wave
  // (weight staging amortized 10x vs one-tile-per-wave)
  k_edge_msg<<<1250, 128, 0, stream>>>(edge_attr, eidx, We, be, WM, bM, h, aggr, deg);

  // update MLP + graph pool: persistent waves
  k_update_pool<<<625, 128, 0, stream>>>(h, aggr, deg, WU, bU, batch, pool, gcnt);

  // final head
  k_final<<<1, 128, 0, stream>>>(pool, gcnt, Wh, bh, out);
}
